// SE3_ResNet_26139170964134
// MI455X (gfx1250) — matrix operand order repaired and checked by reading, not compiled, not run on hardware
//
#include <hip/hip_runtime.h>
#include <hip/hip_bf16.h>
#include <cstdint>
#include <cstddef>

#define NNODES 65536

typedef __attribute__((ext_vector_type(16))) _Float16 v16h;
typedef __attribute__((ext_vector_type(8)))  _Float16 v8h;
typedef __attribute__((ext_vector_type(8)))  float    v8f;

// ---------------------------------------------------------------------------
// Edge radial basis: edge e = (i, i+1), rings at {0, w}, sigma = w/2.
// eb[2*i+r] = exp(-0.5*((r_e - ring_r)/sigma)^2), eb[2*(N-1)+*] = 0.
// ---------------------------------------------------------------------------
__global__ void radial_kernel(const float* __restrict__ x,
                              float* __restrict__ eb1,
                              float* __restrict__ eb10, int Nn) {
  int i = blockIdx.x * blockDim.x + threadIdx.x;
  if (i >= Nn) return;
  float a0 = 0.f, a1 = 0.f, b0 = 0.f, b1 = 0.f;
  if (i < Nn - 1) {
    float dx = x[3 * (i + 1) + 0] - x[3 * i + 0];
    float dy = x[3 * (i + 1) + 1] - x[3 * i + 1];
    float dz = x[3 * (i + 1) + 2] - x[3 * i + 2];
    float r = sqrtf(dx * dx + dy * dy + dz * dz);
    { float s = 0.5f; float t0 = r / s, t1 = (r - 1.0f) / s;
      a0 = expf(-0.5f * t0 * t0); a1 = expf(-0.5f * t1 * t1); }
    { float s = 5.0f; float t0 = r / s, t1 = (r - 10.0f) / s;
      b0 = expf(-0.5f * t0 * t0); b1 = expf(-0.5f * t1 * t1); }
  }
  eb1[2 * i] = a0; eb1[2 * i + 1] = a1;
  eb10[2 * i] = b0; eb10[2 * i + 1] = b1;
}

// ---------------------------------------------------------------------------
// Pack W [2, Cout, Cin] fp32 into WMMA B fragments (f16). K padded per-ring to
// KH (mult of 16), total K = 2*KH (mult of 32). B (32x16) lane layout:
// lane<16: n=lane, K=slot; lane>=16: n=lane-16, K=16+slot.
// Linear: [(kt*NT+nt)*512 + lane*16 + slot].
// ---------------------------------------------------------------------------
__global__ void pack_w_kernel(const float* __restrict__ W,
                              _Float16* __restrict__ Bp,
                              int Cin, int Cout, int KH, int NT, int total) {
  int idx = blockIdx.x * blockDim.x + threadIdx.x;
  if (idx >= total) return;
  int tile = idx >> 9;
  int within = idx & 511;
  int lane = within >> 4;
  int slot = within & 15;
  int nt = tile % NT;
  int kt = tile / NT;
  int nloc = lane & 15;
  int kloc = ((lane >> 4) << 4) + slot;
  int n = nt * 16 + nloc;
  int k = kt * 32 + kloc;
  float v = 0.f;
  if (n < Cout) {
    int ring = (k >= KH) ? 1 : 0;
    int kc = k - ring * KH;
    if (kc < Cin) v = W[((size_t)ring * Cout + n) * Cin + kc];
  }
  Bp[idx] = (_Float16)v;
}

// ---------------------------------------------------------------------------
// Input-side ring stencil: G[i, r*KH + k] =
//   eb[i][r] * F[i+1, k] + eb[i-1][r] * F[i-1, k]   (f16, zero padded)
// ---------------------------------------------------------------------------
__global__ void stencil_in_kernel(const float* __restrict__ F,
                                  const float* __restrict__ eb,
                                  _Float16* __restrict__ G,
                                  int Nn, int Cin, int KH) {
  int idx = blockIdx.x * blockDim.x + threadIdx.x;
  if (idx >= Nn * KH) return;
  int i = idx / KH;
  int k = idx - i * KH;
  float g0 = 0.f, g1 = 0.f;
  if (k < Cin) {
    if (i < Nn - 1) {
      float fr = F[(size_t)(i + 1) * Cin + k];
      g0 += eb[2 * i] * fr;
      g1 += eb[2 * i + 1] * fr;
    }
    if (i > 0) {
      float fl = F[(size_t)(i - 1) * Cin + k];
      g0 += eb[2 * (i - 1)] * fl;
      g1 += eb[2 * (i - 1) + 1] * fl;
    }
  }
  size_t base = (size_t)i * (2 * KH);
  G[base + k] = (_Float16)g0;
  G[base + KH + k] = (_Float16)g1;
}

// ---------------------------------------------------------------------------
// WMMA GEMM: Y[N, Cout] = G[N, Ktot](f16) @ Bpack (+ Res).
// One wave = 16 rows x JT 16-col tiles, JT fixed at compile time so the
// K-loop is branch-free straight-line WMMA (no exec masking, no acc copies).
// ---------------------------------------------------------------------------
template <int JT, bool HASRES>
__global__ __launch_bounds__(256) void gemm_wmma_kernel(
    const _Float16* __restrict__ G, const _Float16* __restrict__ Bp,
    const float* __restrict__ Res, float* __restrict__ Y,
    int Ktot, int Cout, int NT, int nt0base, int NTG, int totalWaves) {
  int w = blockIdx.x * (blockDim.x >> 5) + (threadIdx.x >> 5);
  if (w >= totalWaves) return;  // wave-uniform
  int lane = threadIdx.x & 31;
  int ntg = w % NTG;
  int mt = w / NTG;
  int m0 = mt << 4;
  int nt0 = nt0base + (ntg << 2);

  v8f acc[JT];
#pragma unroll
  for (int j = 0; j < JT; ++j) acc[j] = v8f{};

  int KT = Ktot >> 5;
  int half = lane >> 4;
  const _Float16* grow = G + (size_t)(m0 + (lane & 15)) * Ktot;
  const _Float16* bcol = Bp + (size_t)nt0 * 512 + lane * 16;
  size_t bstride = (size_t)NT * 512;  // per-kt stride

#pragma unroll 2
  for (int kt = 0; kt < KT; ++kt) {
    // A fragment (16x32): lanes0-15 K=[0..15], lanes16-31 K=[16..31]
    v8h lo = *(const v8h*)(grow + kt * 32 + half * 16);
    v8h hi = *(const v8h*)(grow + kt * 32 + half * 16 + 8);
    v16h a = __builtin_shufflevector(lo, hi, 0, 1, 2, 3, 4, 5, 6, 7,
                                     8, 9, 10, 11, 12, 13, 14, 15);
    v16h b[JT];
    const _Float16* bb = bcol + (size_t)kt * bstride;
#pragma unroll
    for (int j = 0; j < JT; ++j) b[j] = *(const v16h*)(bb + (size_t)j * 512);
#pragma unroll
    for (int j = 0; j < JT; ++j)
      acc[j] = __builtin_amdgcn_wmma_f32_16x16x32_f16(
          false, a, false, b[j], (short)0, acc[j], false, false);
  }

  // D layout: lane<16 -> (M=v, N=lane); lane>=16 -> (M=v+8, N=lane-16)
  int nloc = lane & 15;
  int mrow = m0 + half * 8;
#pragma unroll
  for (int j = 0; j < JT; ++j) {
    int n = (nt0 + j) * 16 + nloc;
    if (n < Cout) {
#pragma unroll
      for (int v = 0; v < 8; ++v) {
        size_t idx = (size_t)(mrow + v) * Cout + n;
        float val = acc[j][v];
        if constexpr (HASRES) val += Res[idx];
        Y[idx] = val;
      }
    }
  }
}

// ---------------------------------------------------------------------------
// Field gating: Yg[i, f*40+j] = Yraw[i, f*40+j] * sigmoid(Yraw[i, nf*40+f])
// ---------------------------------------------------------------------------
__global__ void gate_kernel(const float* __restrict__ Yraw,
                            float* __restrict__ Yg, int Nn, int nf) {
  int C = nf * 40;
  int idx = blockIdx.x * blockDim.x + threadIdx.x;
  if (idx >= Nn * C) return;
  int i = idx / C;
  int c = idx - i * C;
  int f = c / 40;
  size_t rb = (size_t)i * (nf * 41);
  float g = Yraw[rb + C + f];
  float s = 1.0f / (1.0f + expf(-g));
  Yg[idx] = Yraw[rb + c] * s;
}

// ---------------------------------------------------------------------------
// NormPool: out[i, f] = || F[i, f*40 : f*40+40] ||_2  (F width 320, 8 fields)
// ---------------------------------------------------------------------------
__global__ void norm_kernel(const float* __restrict__ F,
                            float* __restrict__ out, int Nn) {
  int idx = blockIdx.x * blockDim.x + threadIdx.x;
  if (idx >= Nn * 8) return;
  int i = idx >> 3;
  int f = idx & 7;
  const float* p = F + (size_t)i * 320 + f * 40;
  float s = 0.f;
#pragma unroll
  for (int j = 0; j < 40; ++j) { float v = p[j]; s += v * v; }
  out[idx] = sqrtf(s);
}

// ---------------------------------------------------------------------------
struct Cfg { int Cin, Cout, KH; };  // KH = ceil16(Cin); Ktot = 2*KH
static const Cfg g_cfg[13] = {
    {3, 39, 16},                                      // W0
    {39, 82, 48}, {80, 82, 80}, {39, 80, 48},         // b1: W1, W2, Ws
    {80, 246, 80}, {240, 246, 240}, {80, 240, 80},    // b2
    {240, 492, 240}, {480, 492, 480}, {240, 480, 240},// b3
    {480, 328, 480}, {320, 328, 320}, {480, 320, 480} // b4
};

static void launch_gemm(const _Float16* G, const _Float16* Bp,
                        const float* Res, float* Y, int Ktot, int Cout,
                        hipStream_t stream) {
  int NT = (Cout + 15) >> 4;
  int MT = NNODES >> 4;
  int NTGfull = NT >> 2;
  int rem = NT & 3;
  if (NTGfull > 0) {
    int totalWaves = MT * NTGfull;
    int blocks = (totalWaves + 7) / 8;
    if (Res)
      gemm_wmma_kernel<4, true><<<blocks, 256, 0, stream>>>(
          G, Bp, Res, Y, Ktot, Cout, NT, 0, NTGfull, totalWaves);
    else
      gemm_wmma_kernel<4, false><<<blocks, 256, 0, stream>>>(
          G, Bp, Res, Y, Ktot, Cout, NT, 0, NTGfull, totalWaves);
  }
  if (rem) {
    int totalWaves = MT;
    int blocks = (totalWaves + 7) / 8;
    int nt0base = NTGfull << 2;
    if (Res) {
      switch (rem) {
        case 1: gemm_wmma_kernel<1, true><<<blocks, 256, 0, stream>>>(
                    G, Bp, Res, Y, Ktot, Cout, NT, nt0base, 1, totalWaves); break;
        case 2: gemm_wmma_kernel<2, true><<<blocks, 256, 0, stream>>>(
                    G, Bp, Res, Y, Ktot, Cout, NT, nt0base, 1, totalWaves); break;
        default: gemm_wmma_kernel<3, true><<<blocks, 256, 0, stream>>>(
                    G, Bp, Res, Y, Ktot, Cout, NT, nt0base, 1, totalWaves); break;
      }
    } else {
      switch (rem) {
        case 1: gemm_wmma_kernel<1, false><<<blocks, 256, 0, stream>>>(
                    G, Bp, Res, Y, Ktot, Cout, NT, nt0base, 1, totalWaves); break;
        case 2: gemm_wmma_kernel<2, false><<<blocks, 256, 0, stream>>>(
                    G, Bp, Res, Y, Ktot, Cout, NT, nt0base, 1, totalWaves); break;
        default: gemm_wmma_kernel<3, false><<<blocks, 256, 0, stream>>>(
                    G, Bp, Res, Y, Ktot, Cout, NT, nt0base, 1, totalWaves); break;
      }
    }
  }
}

extern "C" void kernel_launch(void* const* d_in, const int* in_sizes, int n_in,
                              void* d_out, int out_size, void* d_ws,
                              size_t ws_size, hipStream_t stream) {
  (void)in_sizes; (void)n_in; (void)out_size; (void)ws_size;
  const int Nn = NNODES;
  const float* x = (const float*)d_in[0];

  char* ws = (char*)d_ws;
  size_t off = 0;
  auto take = [&](size_t bytes) -> void* {
    void* p = ws + off;
    off += (bytes + 255) & ~(size_t)255;
    return p;
  };
  float* eb1 = (float*)take((size_t)Nn * 2 * sizeof(float));
  float* eb10 = (float*)take((size_t)Nn * 2 * sizeof(float));
  float* actA = (float*)take((size_t)Nn * 512 * sizeof(float));
  float* actB = (float*)take((size_t)Nn * 512 * sizeof(float));
  float* actC = (float*)take((size_t)Nn * 512 * sizeof(float));
  _Float16* G = (_Float16*)take((size_t)Nn * 960 * sizeof(_Float16));
  _Float16* wp[13];
  for (int c = 0; c < 13; ++c) {
    int KT = (2 * g_cfg[c].KH) >> 5;
    int NT = (g_cfg[c].Cout + 15) >> 4;
    wp[c] = (_Float16*)take((size_t)KT * NT * 512 * sizeof(_Float16));
  }

  radial_kernel<<<(Nn + 255) / 256, 256, 0, stream>>>(x, eb1, eb10, Nn);

  for (int c = 0; c < 13; ++c) {
    int KT = (2 * g_cfg[c].KH) >> 5;
    int NT = (g_cfg[c].Cout + 15) >> 4;
    int total = KT * NT * 512;
    pack_w_kernel<<<(total + 255) / 256, 256, 0, stream>>>(
        (const float*)d_in[c + 1], wp[c], g_cfg[c].Cin, g_cfg[c].Cout,
        g_cfg[c].KH, NT, total);
  }

  auto conv = [&](const float* Fin, int c, const float* eb, const float* res,
                  float* Yout) {
    int Cin = g_cfg[c].Cin, Cout = g_cfg[c].Cout, KH = g_cfg[c].KH;
    int Ktot = 2 * KH;
    int tot = Nn * KH;
    stencil_in_kernel<<<(tot + 255) / 256, 256, 0, stream>>>(Fin, eb, G, Nn,
                                                             Cin, KH);
    launch_gemm(G, wp[c], res, Yout, Ktot, Cout, stream);
  };

  // f = conv(x, W0, rb1)
  conv(x, 0, eb1, nullptr, actA);
  float* f = actA;
  float* t1 = actB;
  float* t2 = actC;
  const int nfs[4] = {2, 6, 12, 8};
  for (int b = 0; b < 4; ++b) {
    int c1 = 1 + 3 * b;
    int nf = nfs[b];
    int gtot = Nn * nf * 40;
    conv(f, c1 + 0, eb1, nullptr, t1);                                    // raw1
    gate_kernel<<<(gtot + 255) / 256, 256, 0, stream>>>(t1, t2, Nn, nf);  // y1
    conv(t2, c1 + 1, eb1, nullptr, t1);                                   // raw2
    gate_kernel<<<(gtot + 255) / 256, 256, 0, stream>>>(t1, t2, Nn, nf);  // y2
    conv(f, c1 + 2, eb10, t2, t1);  // f_new = skip-conv + y2 (fused residual)
    float* tmp = f; f = t1; t1 = tmp;
  }
  norm_kernel<<<(Nn * 8 + 255) / 256, 256, 0, stream>>>(f, (float*)d_out, Nn);
}
